// FFFLayer_84859963834977
// MI455X (gfx1250) — matrix operand order repaired and checked by reading, not compiled, not run on hardware
//
#include <hip/hip_runtime.h>

// ---------------------------------------------------------------------------
// FFF tree layer for MI455X (gfx1250), bf16 WMMA (v_wmma_f32_16x16x32_bf16)
// with double-buffered GLOBAL_LOAD_ASYNC_TO_LDS_B128 staging (ASYNCcnt path).
//
// Pipeline:
//  K_cvt_x    : x (f32 4096x768)            -> xb  (bf16, row-major)
//  K_cvt_nw1  : node_w1 (63,768,16)         -> nw1t (bf16 [j=1024 pad][k=768]) + nb1p
//  K_cvt_w1   : leaf_w1 (64,768,128)        -> w1t  (bf16 [n=8192][k=768])
//  K_cvt_w2   : leaf_w2/leaf_b2             -> w2t  (bf16 [o=768][k=8256])  (bias = K rows 8192..8255)
//  gemm<0>    : hn = relu(xb @ nw1t^T + nb1p)        M=4096 N=1024 K=768   (f32 out)
//  K_node_head: c=sigmoid(hn.nw2+nb2); tree product -> w (f32 4096x64) + whl tail (bf16 w)
//  gemm<1>    : whl = bf16( w[b,l] * relu(xb @ w1t^T + leaf_b1) )  M=4096 N=8192 K=768
//  gemm<2>    : out = whl(4096x8256) @ w2t^T          M=4096 N=768 K=8256  (f32 out)
// ---------------------------------------------------------------------------

typedef __attribute__((ext_vector_type(16))) __bf16 v16bf;
typedef __attribute__((ext_vector_type(8)))  float  v8f;

union FragU { v16bf v; uint4 q[2]; };

__device__ __forceinline__ unsigned short f2bf(float f) {
    unsigned int u = __float_as_uint(f);
    u += 0x7FFFu + ((u >> 16) & 1u);          // round-to-nearest-even
    return (unsigned short)(u >> 16);
}

// ---------------- sizes ----------------
#define BATCH   4096
#define IN_DIM  768
#define N_NODES 63
#define NODE_N  1024      /* padded 63*16 -> 1024 */
#define LEAF_N  8192      /* 64*128 */
#define K_EXT   8256      /* 8192 + 64 (bias fold) */
#define OUT_DIM 768

// ---------------- conversion / transpose kernels ----------------
__global__ void K_cvt_x(const float* __restrict__ x, unsigned short* __restrict__ xb) {
    int i = blockIdx.x * blockDim.x + threadIdx.x;           // one thread = 4 elems
    if (i >= (BATCH * IN_DIM) / 4) return;
    float4 v = ((const float4*)x)[i];
    ushort4 o;
    o.x = f2bf(v.x); o.y = f2bf(v.y); o.z = f2bf(v.z); o.w = f2bf(v.w);
    ((ushort4*)xb)[i] = o;
}

__global__ void K_cvt_nw1(const float* __restrict__ nw1, const float* __restrict__ nb1,
                          unsigned short* __restrict__ nw1t, float* __restrict__ nb1p) {
    int idx = blockIdx.x * blockDim.x + threadIdx.x;         // over 1024*768
    if (idx >= NODE_N * IN_DIM) return;
    int j = idx / IN_DIM, k = idx - j * IN_DIM;
    float v = 0.f;
    if (j < N_NODES * 16) v = nw1[(size_t)(j >> 4) * (IN_DIM * 16) + (size_t)k * 16 + (j & 15)];
    nw1t[idx] = f2bf(v);
    if (idx < NODE_N) nb1p[idx] = (idx < N_NODES * 16) ? nb1[idx] : 0.f;
}

__global__ void K_cvt_w1(const float* __restrict__ w1, unsigned short* __restrict__ w1t) {
    int idx = blockIdx.x * blockDim.x + threadIdx.x;         // over 8192*768
    if (idx >= LEAF_N * IN_DIM) return;
    int n = idx / IN_DIM, k = idx - n * IN_DIM;
    w1t[idx] = f2bf(w1[(size_t)(n >> 7) * (IN_DIM * 128) + (size_t)k * 128 + (n & 127)]);
}

__global__ void K_cvt_w2(const float* __restrict__ w2, const float* __restrict__ b2,
                         unsigned short* __restrict__ w2t) {
    int idx = blockIdx.x * blockDim.x + threadIdx.x;         // over 768*8256
    if (idx >= OUT_DIM * K_EXT) return;
    int o = idx / K_EXT, k = idx - o * K_EXT;
    float v = (k < LEAF_N) ? w2[(size_t)k * OUT_DIM + o]
                           : b2[(size_t)(k - LEAF_N) * OUT_DIM + o];
    w2t[idx] = f2bf(v);
}

// ---------------- node head: sigmoid gates + depth-6 soft routing ----------------
__global__ __launch_bounds__(64) void K_node_head(
    const float* __restrict__ hn, const float* __restrict__ nw2, const float* __restrict__ nb2,
    float* __restrict__ wbuf, unsigned short* __restrict__ whl) {
    __shared__ float cc[64];
    int b = blockIdx.x, t = threadIdx.x;
    if (t < N_NODES) {
        const float* hr = hn + (size_t)b * NODE_N + t * 16;
        float s = nb2[t];
        #pragma unroll
        for (int h = 0; h < 16; ++h) s += hr[h] * nw2[t * 16 + h];
        cc[t] = 1.f / (1.f + __expf(-s));
    }
    __syncthreads();
    float w = 1.f;
    #pragma unroll
    for (int lvl = 0; lvl < 6; ++lvl) {
        int n   = (1 << lvl) - 1 + (t >> (6 - lvl));
        int bit = (t >> (5 - lvl)) & 1;
        float g = cc[n];
        w *= bit ? (1.f - g) : g;
    }
    wbuf[(size_t)b * 64 + t] = w;
    whl[(size_t)b * K_EXT + LEAF_N + t] = f2bf(w);           // bias-fold tail columns
}

// ---------------- tiled bf16 WMMA GEMM, async-LDS double buffered ----------------
// C(MxN) = A(MxK) * Bt(NxK)^T ; BM=BN=128, BK=32, 256 threads = 8 waves (4x2),
// wave tile 32x64 -> 2x4 = 8 x v_wmma_f32_16x16x32_bf16 per K step.
// Staging: per-lane GLOBAL_LOAD_ASYNC_TO_LDS_B128 (ASYNCcnt), 2 LDS buffers,
// one barrier per K step (syncthreads => s_wait_dscnt 0 + s_barrier).
// MODE 0: f32 out, +bias[n], relu            (hn)
// MODE 1: bf16 out, +bias[n], relu, *w[row]  (whl)
// MODE 2: f32 out, raw                       (final out)
#define LDSROW 40   /* bf16 elems per LDS row (32 + pad) -> 80B = 20 banks, conflict-free */

__device__ __forceinline__ void async_b128(unsigned lds_addr, const unsigned short* gptr) {
    asm volatile("global_load_async_to_lds_b128 %0, %1, off"
                 :: "v"(lds_addr), "v"(gptr) : "memory");
}
__device__ __forceinline__ void wait_async0() {
    asm volatile("s_wait_asynccnt 0x0" ::: "memory");
}

template <int MODE>
__global__ __launch_bounds__(256) void gemm_bf16(
    const unsigned short* __restrict__ A, const unsigned short* __restrict__ Bt,
    void* __restrict__ Cv, const float* __restrict__ bias,
    const float* __restrict__ wleaf, int K, int ldc) {

    __shared__ unsigned short As[2][128 * LDSROW];
    __shared__ unsigned short Bs[2][128 * LDSROW];
    __shared__ float sw[128];

    const int t  = threadIdx.x;
    const int ln = t & 31, wv = t >> 5;
    const int lo = ln & 15, hi = ln >> 4;
    const int m0 = blockIdx.y * 128, n0 = blockIdx.x * 128;
    const int mw = (wv & 3) * 32, nw = (wv >> 2) * 64;

    if (MODE == 1) {                                          // per-row leaf gate
        if (t < 128) sw[t] = wleaf[(size_t)(m0 + t) * 64 + (n0 >> 7)];
    }

    const int sr = t >> 2;        // staging row 0..63
    const int sq = t & 3;         // staging 16B quarter

    // per-lane LDS destinations for the 4 async b128 chunks, both buffers
    unsigned ldsA0[2], ldsA1[2], ldsB0[2], ldsB1[2];
    #pragma unroll
    for (int bfi = 0; bfi < 2; ++bfi) {
        ldsA0[bfi] = (unsigned)(size_t)&As[bfi][sr * LDSROW + sq * 8];
        ldsA1[bfi] = (unsigned)(size_t)&As[bfi][(sr + 64) * LDSROW + sq * 8];
        ldsB0[bfi] = (unsigned)(size_t)&Bs[bfi][sr * LDSROW + sq * 8];
        ldsB1[bfi] = (unsigned)(size_t)&Bs[bfi][(sr + 64) * LDSROW + sq * 8];
    }
    const unsigned short* gA0 = A  + (size_t)(m0 + sr)      * K + sq * 8;
    const unsigned short* gA1 = A  + (size_t)(m0 + sr + 64) * K + sq * 8;
    const unsigned short* gB0 = Bt + (size_t)(n0 + sr)      * K + sq * 8;
    const unsigned short* gB1 = Bt + (size_t)(n0 + sr + 64) * K + sq * 8;

    int aoff[2], boff[4];
    #pragma unroll
    for (int i = 0; i < 2; ++i) aoff[i] = (mw + i * 16 + lo) * LDSROW + hi * 8;
    #pragma unroll
    for (int j = 0; j < 4; ++j) boff[j] = (nw + j * 16 + lo) * LDSROW + hi * 8;

    const v8f vzero = {0.f, 0.f, 0.f, 0.f, 0.f, 0.f, 0.f, 0.f};
    v8f acc[2][4];
    #pragma unroll
    for (int i = 0; i < 2; ++i)
        #pragma unroll
        for (int j = 0; j < 4; ++j) acc[i][j] = vzero;

    // prologue: async-stage K tile 0 into buffer 0
    async_b128(ldsA0[0], gA0);
    async_b128(ldsA1[0], gA1);
    async_b128(ldsB0[0], gB0);
    async_b128(ldsB1[0], gB1);

    const int nk = K >> 5;
    for (int ik = 0; ik < nk; ++ik) {
        const int buf = ik & 1;
        wait_async0();            // own async loads into `buf` complete
        __syncthreads();          // emits s_wait_dscnt 0 + barrier: prev frag reads done

        if (ik + 1 < nk) {        // async-stage next K tile into other buffer
            const int kq = (ik + 1) << 5;
            async_b128(ldsA0[buf ^ 1], gA0 + kq);
            async_b128(ldsA1[buf ^ 1], gA1 + kq);
            async_b128(ldsB0[buf ^ 1], gB0 + kq);
            async_b128(ldsB1[buf ^ 1], gB1 + kq);
        }

        // ---- fragment loads per ISA VGPR layouts (16-bit A 16x32 / B 32x16) ----
        FragU a[2], b[4];
        #pragma unroll
        for (int i = 0; i < 2; ++i) {
            a[i].q[0] = *(const uint4*)&As[buf][aoff[i]];     // K = 8*hi   .. +7
            a[i].q[1] = *(const uint4*)&As[buf][aoff[i] + 16];// K = 8*hi+16.. +23
        }
        #pragma unroll
        for (int j = 0; j < 4; ++j) {
            b[j].q[0] = *(const uint4*)&Bs[buf][boff[j]];     // K = 8*hi   .. +7
            b[j].q[1] = *(const uint4*)&Bs[buf][boff[j] + 16];// K = 8*hi+16.. +23
        }

        #pragma unroll
        for (int i = 0; i < 2; ++i)
            #pragma unroll
            for (int j = 0; j < 4; ++j)
                acc[i][j] = __builtin_amdgcn_wmma_f32_16x16x32_bf16(
                    false, a[i].v, false, b[j].v, (short)0, acc[i][j], false, false);
    }

    // ---- epilogue: C/D layout => lane: N = lo, M = r + 8*hi (per 16x16 tile) ----
    float* Cf = (float*)Cv;
    unsigned short* Cb = (unsigned short*)Cv;
    #pragma unroll
    for (int i = 0; i < 2; ++i) {
        const int mloc = mw + i * 16 + hi * 8;
        #pragma unroll
        for (int j = 0; j < 4; ++j) {
            const int n = n0 + nw + j * 16 + lo;
            float bia = 0.f;
            if (MODE != 2) bia = bias[n];
            #pragma unroll
            for (int r = 0; r < 8; ++r) {
                const size_t row = (size_t)(m0 + mloc + r);
                float v = acc[i][j][r];
                if (MODE != 2) { v += bia; v = v > 0.f ? v : 0.f; }
                if (MODE == 0)      Cf[row * ldc + n] = v;
                else if (MODE == 1) Cb[row * ldc + n] = f2bf(v * sw[mloc + r]);
                else                Cf[row * ldc + n] = v;
            }
        }
    }
}

// ---------------- workspace layout (bytes) ----------------
static constexpr size_t OFF_XB   = 0;                                          // 4096x768  bf16
static constexpr size_t OFF_NW1T = OFF_XB   + (size_t)BATCH * IN_DIM * 2;      // 1024x768 bf16
static constexpr size_t OFF_NB1P = OFF_NW1T + (size_t)NODE_N * IN_DIM * 2;     // 1024 f32
static constexpr size_t OFF_HN   = OFF_NB1P + (size_t)NODE_N * 4;              // 4096x1024 f32
static constexpr size_t OFF_W    = OFF_HN   + (size_t)BATCH * NODE_N * 4;      // 4096x64 f32
static constexpr size_t OFF_W1T  = OFF_W    + (size_t)BATCH * 64 * 4;          // 8192x768 bf16
static constexpr size_t OFF_W2T  = OFF_W1T  + (size_t)LEAF_N * IN_DIM * 2;     // 768x8256 bf16
static constexpr size_t OFF_WHL  = OFF_W2T  + (size_t)OUT_DIM * K_EXT * 2;     // 4096x8256 bf16

extern "C" void kernel_launch(void* const* d_in, const int* in_sizes, int n_in,
                              void* d_out, int out_size, void* d_ws, size_t ws_size,
                              hipStream_t stream) {
    const float* x   = (const float*)d_in[0];
    const float* nw1 = (const float*)d_in[1];
    const float* nb1 = (const float*)d_in[2];
    const float* nw2 = (const float*)d_in[3];
    const float* nb2 = (const float*)d_in[4];
    const float* lw1 = (const float*)d_in[5];
    const float* lb1 = (const float*)d_in[6];
    const float* lw2 = (const float*)d_in[7];
    const float* lb2 = (const float*)d_in[8];
    float* out = (float*)d_out;

    char* ws = (char*)d_ws;
    unsigned short* xb   = (unsigned short*)(ws + OFF_XB);
    unsigned short* nw1t = (unsigned short*)(ws + OFF_NW1T);
    float*          nb1p = (float*)         (ws + OFF_NB1P);
    float*          hn   = (float*)         (ws + OFF_HN);
    float*          wbuf = (float*)         (ws + OFF_W);
    unsigned short* w1t  = (unsigned short*)(ws + OFF_W1T);
    unsigned short* w2t  = (unsigned short*)(ws + OFF_W2T);
    unsigned short* whl  = (unsigned short*)(ws + OFF_WHL);

    const int TB = 256;
    // conversions / transposes
    K_cvt_x  <<<(BATCH * IN_DIM / 4 + TB - 1) / TB, TB, 0, stream>>>(x, xb);
    K_cvt_nw1<<<(NODE_N * IN_DIM + TB - 1) / TB, TB, 0, stream>>>(nw1, nb1, nw1t, nb1p);
    K_cvt_w1 <<<(LEAF_N * IN_DIM + TB - 1) / TB, TB, 0, stream>>>(lw1, w1t);
    K_cvt_w2 <<<(OUT_DIM * K_EXT + TB - 1) / TB, TB, 0, stream>>>(lw2, lb2, w2t);

    // node GEMM: hn = relu(xb @ nw1t^T + nb1p)   M=4096 N=1024 K=768
    gemm_bf16<0><<<dim3(NODE_N / 128, BATCH / 128), TB, 0, stream>>>(
        xb, nw1t, hn, nb1p, nullptr, IN_DIM, NODE_N);

    // gates + soft routing weights
    K_node_head<<<BATCH, 64, 0, stream>>>(hn, nw2, nb2, wbuf, whl);

    // leaf GEMM 1: whl = bf16(w * relu(xb @ w1t^T + lb1))   M=4096 N=8192 K=768
    gemm_bf16<1><<<dim3(LEAF_N / 128, BATCH / 128), TB, 0, stream>>>(
        xb, w1t, whl, lb1, wbuf, IN_DIM, K_EXT);

    // leaf GEMM 2 (+bias via K-extension): out = whl @ w2t^T   M=4096 N=768 K=8256
    gemm_bf16<2><<<dim3(OUT_DIM / 128, BATCH / 128), TB, 0, stream>>>(
        whl, w2t, out, nullptr, nullptr, K_EXT, OUT_DIM);
}
